// DecoderBlock_20512763806056
// MI455X (gfx1250) — matrix operand order repaired and checked by reading, not compiled, not run on hardware
//
#include <hip/hip_runtime.h>

typedef __attribute__((ext_vector_type(16))) __bf16 v16bf;
typedef __attribute__((ext_vector_type(8)))  float  v8f;

union FragAB { uint4 u[2]; v16bf v; };
static_assert(sizeof(v16bf) == 32, "v16bf size");

#define FLAG_F32OUT 1
#define FLAG_TSTORE 2
#define FLAG_GELU   4

__device__ __forceinline__ unsigned short f2bf(float f) {
    unsigned int u = __float_as_uint(f);
    u += 0x7FFFu + ((u >> 16) & 1u);           // round-to-nearest-even
    return (unsigned short)(u >> 16);
}

__device__ __forceinline__ float gelu_t(float x) {
    float x3 = x * x * x;
    return 0.5f * x * (1.f + tanhf(0.7978845608028654f * (x + 0.044715f * x3)));
}

__device__ __forceinline__ v8f wmma_bf(const FragAB& a, const FragAB& b, v8f c) {
    return __builtin_amdgcn_wmma_f32_16x16x32_bf16(false, a.v, false, b.v,
                                                   (short)0, c, false, false);
}

// ---------------- weight transpose/convert ----------------
// in: fp32 [K,N] row-major -> out: bf16 [N,K] row-major
__global__ void t_mat(const float* __restrict__ in, unsigned short* __restrict__ out,
                      int K, int N) {
    long long i = (long long)blockIdx.x * blockDim.x + threadIdx.x;
    long long tot = (long long)K * N;
    if (i >= tot) return;
    int n = (int)(i / K);
    int k = (int)(i % K);
    out[i] = f2bf(in[(long long)k * N + n]);
}

// wq [H=12, D=768, HD=64] fp32 -> out bf16 [N=768 (h*64+hd), K=768 (d)]
__global__ void t_qkv(const float* __restrict__ w, unsigned short* __restrict__ out) {
    int i = blockIdx.x * blockDim.x + threadIdx.x;
    if (i >= 768 * 768) return;
    int n = i / 768, d = i % 768;
    int h = n >> 6, hd = n & 63;
    out[i] = f2bf(w[((long long)h * 768 + d) * 64 + hd]);
}

// ---------------- LayerNorm -> bf16 ----------------
__global__ __launch_bounds__(256) void ln_bf16(const float* __restrict__ x,
                                               const float* __restrict__ g,
                                               const float* __restrict__ be,
                                               unsigned short* __restrict__ out,
                                               int rows) {
    int wid = threadIdx.x >> 5, lane = threadIdx.x & 31;
    int row = blockIdx.x * 8 + wid;
    if (row >= rows) return;
    const float* xr = x + (size_t)row * 768;
    float s = 0.f, s2 = 0.f;
#pragma unroll
    for (int i = 0; i < 24; ++i) {
        float v = xr[lane + i * 32];
        s += v; s2 += v * v;
    }
#pragma unroll
    for (int m = 1; m < 32; m <<= 1) {
        s  += __shfl_xor(s,  m, 32);
        s2 += __shfl_xor(s2, m, 32);
    }
    float mu  = s  * (1.f / 768.f);
    float var = s2 * (1.f / 768.f) - mu * mu;
    float rs  = rsqrtf(var + 1e-5f);
    unsigned short* orow = out + (size_t)row * 768;
#pragma unroll
    for (int i = 0; i < 24; ++i) {
        int c = lane + i * 32;
        orow[c] = f2bf((xr[c] - mu) * rs * g[c] + be[c]);
    }
}

// ---------------- register-blocked bf16 WMMA GEMM ----------------
// Each wave computes a 32(M) x 64(N) output tile: 2x4 grid of 16x16 WMMA tiles.
// C[M,N] = A[M,K](bf16, lda) * Bt[N,K](bf16)^T + bias[N]  (+resid, gelu, store modes)
__global__ __launch_bounds__(256) void gemm_bf16(
    const unsigned short* __restrict__ A, int lda,
    const unsigned short* __restrict__ Bt,
    const float* __restrict__ bias,
    const float* __restrict__ resid,
    void* __restrict__ out,
    int M, int N, int K, int flags)
{
    int wid = threadIdx.x >> 5, lane = threadIdx.x & 31;
    int Nt4 = N >> 6;                            // 64-col supertiles in N
    int tid = blockIdx.x * 8 + wid;
    int mt = tid / Nt4, nt = tid % Nt4;          // wave-uniform
    if (mt >= (M >> 5)) return;

    int l16 = lane & 15, half = lane >> 4;
    int koff = half ? 16 : 0, loff = half ? 16 : 0;

    const unsigned short* arow0 = A + (size_t)(mt * 32 + l16) * lda;
    const unsigned short* arow1 = arow0 + (size_t)16 * lda;
    const unsigned short* brow  = Bt + (size_t)(nt * 64 + l16) * K;
    const size_t bstride = (size_t)16 * K;       // between N-frags

    v8f acc[2][4];
#pragma unroll
    for (int i = 0; i < 2; ++i)
#pragma unroll
        for (int j = 0; j < 4; ++j)
            acc[i][j] = (v8f){0.f,0.f,0.f,0.f,0.f,0.f,0.f,0.f};

    for (int kb = 0; kb < K; kb += 32) {
        FragAB a0, a1;
        a0.u[0] = *(const uint4*)(arow0 + kb + koff);
        a0.u[1] = *(const uint4*)(arow0 + kb + 8 + koff);
        a1.u[0] = *(const uint4*)(arow1 + kb + koff);
        a1.u[1] = *(const uint4*)(arow1 + kb + 8 + koff);
#pragma unroll
        for (int j = 0; j < 4; ++j) {
            FragAB b;
            const unsigned short* br = brow + (size_t)j * bstride + kb + loff;
            b.u[0] = *(const uint4*)br;
            b.u[1] = *(const uint4*)(br + 8);
            acc[0][j] = wmma_bf(a0, b, acc[0][j]);
            acc[1][j] = wmma_bf(a1, b, acc[1][j]);
        }
    }

#pragma unroll
    for (int i = 0; i < 2; ++i) {
#pragma unroll
        for (int j = 0; j < 4; ++j) {
            int n0 = nt * 64 + j * 16 + l16;
            float bs = bias[n0];
#pragma unroll
            for (int r = 0; r < 8; ++r) {
                int m0 = mt * 32 + i * 16 + r + (half ? 8 : 0);
                float v = acc[i][j][r] + bs;
                if (flags & FLAG_GELU) v = gelu_t(v);
                if (resid) v += resid[(size_t)m0 * N + n0];
                if (flags & FLAG_F32OUT) {
                    ((float*)out)[(size_t)m0 * N + n0] = v;
                } else if (flags & FLAG_TSTORE) {
                    // V output transposed per batch: Vt[b][n0][t], m0 = b*1024 + t
                    size_t idx = (size_t)(m0 >> 10) * ((size_t)N << 10)
                               + (size_t)n0 * 1024 + (size_t)(m0 & 1023);
                    ((unsigned short*)out)[idx] = f2bf(v);
                } else {
                    ((unsigned short*)out)[(size_t)m0 * N + n0] = f2bf(v);
                }
            }
        }
    }
}

// ---------------- fused causal flash attention ----------------
// Qb,Kb: bf16 [B,T,H,HD] ; Vt: bf16 [B,H,HD,T] ; Y: bf16 [B,T,H,HD]
__global__ __launch_bounds__(128) void attn_kernel(
    const unsigned short* __restrict__ Qb,
    const unsigned short* __restrict__ Kb,
    const unsigned short* __restrict__ Vt,
    unsigned short* __restrict__ Y)
{
    __shared__ __align__(16) unsigned short plds[4][16][32];

    int wid = threadIdx.x >> 5, lane = threadIdx.x & 31;
    int tile = blockIdx.x * 4 + wid;            // 6144 tiles total
    int qt = tile & 63;
    int bh = tile >> 6;
    int h = bh % 12, b = bh / 12;
    int qbase = qt * 16;

    int l16 = lane & 15, half = lane >> 4;
    int koff = half ? 16 : 0, loff = half ? 16 : 0;

    // Q A-fragments (16 rows x 64 hd) -> two k-chunks of 32
    FragAB qf0, qf1;
    {
        const unsigned short* qrow =
            Qb + ((size_t)(b * 1024 + qbase + l16) * 768 + h * 64);
        qf0.u[0] = *(const uint4*)(qrow + koff);
        qf0.u[1] = *(const uint4*)(qrow + 8 + koff);
        qf1.u[0] = *(const uint4*)(qrow + 32 + koff);
        qf1.u[1] = *(const uint4*)(qrow + 40 + koff);
    }

    v8f O0 = {0.f,0.f,0.f,0.f,0.f,0.f,0.f,0.f};
    v8f O1 = O0, O2 = O0, O3 = O0;
    float rowmax[8], rowsum[8];
#pragma unroll
    for (int r = 0; r < 8; ++r) { rowmax[r] = -3.0e38f; rowsum[r] = 0.f; }

    unsigned short* L = &plds[wid][0][0];
    const unsigned short* vbase = Vt + ((size_t)(b * 12 + h) * 64) * 1024;
    const float scale = 0.125f;                 // 1/sqrt(64)

    int nkt = ((qbase + 15) >> 5) + 1;          // 32-key tiles needed (causal)
    for (int kt = 0; kt < nkt; ++kt) {
        int kb0 = kt * 32;

        // scores S = Q * K^T  (two 16-key column tiles)
        v8f S0 = {0.f,0.f,0.f,0.f,0.f,0.f,0.f,0.f}, S1 = S0;
        {
            const unsigned short* krow =
                Kb + ((size_t)(b * 1024 + kb0 + l16) * 768 + h * 64);
            FragAB kf;
            kf.u[0] = *(const uint4*)(krow + loff);
            kf.u[1] = *(const uint4*)(krow + loff + 8);
            S0 = wmma_bf(qf0, kf, S0);
            kf.u[0] = *(const uint4*)(krow + 32 + loff);
            kf.u[1] = *(const uint4*)(krow + 40 + loff);
            S0 = wmma_bf(qf1, kf, S0);
            const unsigned short* krow2 = krow + (size_t)16 * 768;
            kf.u[0] = *(const uint4*)(krow2 + loff);
            kf.u[1] = *(const uint4*)(krow2 + loff + 8);
            S1 = wmma_bf(qf0, kf, S1);
            kf.u[0] = *(const uint4*)(krow2 + 32 + loff);
            kf.u[1] = *(const uint4*)(krow2 + 40 + loff);
            S1 = wmma_bf(qf1, kf, S1);
        }

        // online softmax update (per C-layout row M = r + half*8)
        float p0[8], p1[8];
#pragma unroll
        for (int r = 0; r < 8; ++r) {
            int q = qbase + r + (half ? 8 : 0);
            int key0 = kb0 + l16, key1 = key0 + 16;
            float s0 = (key0 <= q) ? S0[r] * scale : -3.0e38f;
            float s1 = (key1 <= q) ? S1[r] * scale : -3.0e38f;
            float t = fmaxf(s0, s1);
            t = fmaxf(t, __shfl_xor(t, 1, 32));
            t = fmaxf(t, __shfl_xor(t, 2, 32));
            t = fmaxf(t, __shfl_xor(t, 4, 32));
            t = fmaxf(t, __shfl_xor(t, 8, 32));
            float mnew = fmaxf(rowmax[r], t);
            float corr = __expf(rowmax[r] - mnew);
            rowmax[r] = mnew;
            float e0 = __expf(s0 - mnew);
            float e1 = __expf(s1 - mnew);
            float ts = e0 + e1;
            ts += __shfl_xor(ts, 1, 32);
            ts += __shfl_xor(ts, 2, 32);
            ts += __shfl_xor(ts, 4, 32);
            ts += __shfl_xor(ts, 8, 32);
            rowsum[r] = rowsum[r] * corr + ts;
            O0[r] *= corr; O1[r] *= corr; O2[r] *= corr; O3[r] *= corr;
            p0[r] = e0; p1[r] = e1;
        }

        // bounce P through LDS: C-layout -> A-fragment layout (16x32 bf16)
#pragma unroll
        for (int r = 0; r < 8; ++r) {
            int m = r + (half ? 8 : 0);
            L[m * 32 + l16]      = f2bf(p0[r]);
            L[m * 32 + l16 + 16] = f2bf(p1[r]);
        }
        FragAB pf;
        {
            const unsigned short* Lr = L + l16 * 32;
            pf.u[0] = *(const uint4*)(Lr + koff);
            pf.u[1] = *(const uint4*)(Lr + 8 + koff);
        }

        // O += P * V   (4 hd chunks of 16)
        {
            const unsigned short* vr0 = vbase + (size_t)(0 * 16 + l16) * 1024 + kb0 + loff;
            const unsigned short* vr1 = vbase + (size_t)(1 * 16 + l16) * 1024 + kb0 + loff;
            const unsigned short* vr2 = vbase + (size_t)(2 * 16 + l16) * 1024 + kb0 + loff;
            const unsigned short* vr3 = vbase + (size_t)(3 * 16 + l16) * 1024 + kb0 + loff;
            FragAB vf;
            vf.u[0] = *(const uint4*)vr0; vf.u[1] = *(const uint4*)(vr0 + 8);
            O0 = wmma_bf(pf, vf, O0);
            vf.u[0] = *(const uint4*)vr1; vf.u[1] = *(const uint4*)(vr1 + 8);
            O1 = wmma_bf(pf, vf, O1);
            vf.u[0] = *(const uint4*)vr2; vf.u[1] = *(const uint4*)(vr2 + 8);
            O2 = wmma_bf(pf, vf, O2);
            vf.u[0] = *(const uint4*)vr3; vf.u[1] = *(const uint4*)(vr3 + 8);
            O3 = wmma_bf(pf, vf, O3);
        }
    }

    // normalize and store Y bf16 [B,T,H,HD]
#pragma unroll
    for (int r = 0; r < 8; ++r) {
        float inv = 1.f / rowsum[r];
        size_t rowoff = (size_t)(b * 1024 + qbase + r + (half ? 8 : 0)) * 768
                      + h * 64 + l16;
        Y[rowoff + 0]  = f2bf(O0[r] * inv);
        Y[rowoff + 16] = f2bf(O1[r] * inv);
        Y[rowoff + 32] = f2bf(O2[r] * inv);
        Y[rowoff + 48] = f2bf(O3[r] * inv);
    }
}

// ---------------- host launch ----------------
extern "C" void kernel_launch(void* const* d_in, const int* in_sizes, int n_in,
                              void* d_out, int out_size, void* d_ws, size_t ws_size,
                              hipStream_t stream) {
    const float* x      = (const float*)d_in[0];
    const float* wq     = (const float*)d_in[1];
    const float* bq     = (const float*)d_in[2];
    const float* wk     = (const float*)d_in[3];
    const float* bk     = (const float*)d_in[4];
    const float* wv     = (const float*)d_in[5];
    const float* bv     = (const float*)d_in[6];
    const float* w_proj = (const float*)d_in[7];
    const float* b_proj = (const float*)d_in[8];
    const float* g1     = (const float*)d_in[9];
    const float* be1    = (const float*)d_in[10];
    const float* g2     = (const float*)d_in[11];
    const float* be2    = (const float*)d_in[12];
    const float* w_fc   = (const float*)d_in[13];
    const float* b_fc   = (const float*)d_in[14];
    const float* w_cp   = (const float*)d_in[15];
    const float* b_cp   = (const float*)d_in[16];

    char* ws = (char*)d_ws;
    auto take = [&](size_t bytes) { char* p = ws; ws += bytes; return p; };
    unsigned short* hb  = (unsigned short*)take(12582912); // [8192,768] bf16
    unsigned short* Qb  = (unsigned short*)take(12582912); // [B,T,H,HD]
    unsigned short* Kb  = (unsigned short*)take(12582912); // [B,T,H,HD]
    unsigned short* Vt  = (unsigned short*)take(12582912); // [B,H,HD,T]
    unsigned short* Yb  = (unsigned short*)take(12582912); // [B,T,D]
    float*          x1  = (float*)take(25165824);          // [8192,768] f32
    unsigned short* h2  = (unsigned short*)take(12582912); // [8192,768]
    unsigned short* fcb = (unsigned short*)take(50331648); // [8192,3072]
    unsigned short* wqT = (unsigned short*)take(1179648);  // [768,768]
    unsigned short* wkT = (unsigned short*)take(1179648);
    unsigned short* wvT = (unsigned short*)take(1179648);
    unsigned short* wpT = (unsigned short*)take(1179648);
    unsigned short* wfT = (unsigned short*)take(4718592);  // [3072,768]
    unsigned short* wcT = (unsigned short*)take(4718592);  // [768,3072]

    // weight conversion (bf16, transposed to [N,K])
    t_qkv<<<2304, 256, 0, stream>>>(wq, wqT);
    t_qkv<<<2304, 256, 0, stream>>>(wk, wkT);
    t_qkv<<<2304, 256, 0, stream>>>(wv, wvT);
    t_mat<<<2304, 256, 0, stream>>>(w_proj, wpT, 768, 768);
    t_mat<<<9216, 256, 0, stream>>>(w_fc,   wfT, 768, 3072);
    t_mat<<<9216, 256, 0, stream>>>(w_cp,   wcT, 3072, 768);

    // LN1
    ln_bf16<<<1024, 256, 0, stream>>>(x, g1, be1, hb, 8192);

    // QKV projections (supertiles: M/32=256, N/64=12 -> 3072 waves -> 384 blocks)
    gemm_bf16<<<384, 256, 0, stream>>>(hb, 768, wqT, bq, nullptr, Qb,
                                       8192, 768, 768, 0);
    gemm_bf16<<<384, 256, 0, stream>>>(hb, 768, wkT, bk, nullptr, Kb,
                                       8192, 768, 768, 0);
    gemm_bf16<<<384, 256, 0, stream>>>(hb, 768, wvT, bv, nullptr, Vt,
                                       8192, 768, 768, FLAG_TSTORE);

    // fused causal attention
    attn_kernel<<<1536, 128, 0, stream>>>(Qb, Kb, Vt, Yb);

    // output projection + residual -> x1 (fp32)
    gemm_bf16<<<384, 256, 0, stream>>>(Yb, 768, wpT, b_proj, x, x1,
                                       8192, 768, 768, FLAG_F32OUT);

    // LN2
    ln_bf16<<<1024, 256, 0, stream>>>(x1, g2, be2, h2, 8192);

    // MLP (fc: 256 x 48 supertiles -> 12288 waves -> 1536 blocks)
    gemm_bf16<<<1536, 256, 0, stream>>>(h2, 768, wfT, b_fc, nullptr, fcb,
                                        8192, 3072, 768, FLAG_GELU);
    gemm_bf16<<<384, 256, 0, stream>>>(fcb, 3072, wcT, b_cp, x1, (float*)d_out,
                                       8192, 768, 3072, FLAG_F32OUT);
}
